// Timedomain_34411277976419
// MI455X (gfx1250) — matrix operand order repaired and checked by reading, not compiled, not run on hardware
//
#include <hip/hip_runtime.h>
#include <hip/hip_bf16.h>

typedef __attribute__((ext_vector_type(16))) __bf16 v16bf;
typedef __attribute__((ext_vector_type(8)))  __bf16 v8bf;
typedef __attribute__((ext_vector_type(8)))  float  v8f;

#define DEV static __device__ __forceinline__

DEV float geluf(float x) { return 0.5f * x * (1.0f + erff(x * 0.70710678118654752f)); }

// ---------------------------------------------------------------- utilities
__global__ void zeroker(float* p, int n) {
    int i = blockIdx.x * 256 + threadIdx.x;
    if (i < n) p[i] = 0.f;
}

// Precompute WMMA A-fragments for W1^T (M=32,K=128) and W2^T (M=128,K=32), bf16.
__global__ void prep_weights(const float* __restrict__ W1, const float* __restrict__ W2,
                             __bf16* __restrict__ w1frag, __bf16* __restrict__ w2frag) {
    int t = threadIdx.x;
    for (int j = 0; j < 16; j++) {           // W1frag: [mt2][kt4][lane32][e16]
        int idx = t + j * 256;
        int e = idx & 15, lane = (idx >> 4) & 31, kt = (idx >> 9) & 3, mt = (idx >> 11) & 1;
        int half = (lane < 16) ? 0 : 8;
        int r = mt * 16 + (lane & 15);                      // f' (output)
        int k = kt * 32 + 2 * half + e;                     // c (input)
        w1frag[idx] = (__bf16)W1[k * 32 + r];               // W1^T[f'][c] = W1[c][f']
    }
    for (int j = 0; j < 16; j++) {           // W2frag: [mt8][lane32][e16]
        int idx = t + j * 256;
        int e = idx & 15, lane = (idx >> 4) & 31, mt = (idx >> 9) & 7;
        int half = (lane < 16) ? 0 : 8;
        int r = mt * 16 + (lane & 15);                      // c (output)
        int k = 2 * half + e;                               // f (input)
        w2frag[idx] = (__bf16)W2[k * 128 + r];              // W2^T[c][f] = W2[f][c]
    }
}

// ------------------------------------------------------- depthwise 3x3 conv
// x: [B][128][128][128] f32.  out: [b][hw][c] bf16 (transposed via LDS).
// stats[c*2 / +1]: atomic sum / sumsq over (B,H,W) for BN2d.
__global__ __launch_bounds__(256) void dwconv(const float* __restrict__ x,
                                              const float* __restrict__ dww,
                                              __bf16* __restrict__ out,
                                              float* __restrict__ stats,
                                              int stride, int So) {
    const int S = 128;
    int tpr = So >> 4;
    int bi = blockIdx.x;
    int ocol0 = (bi % tpr) * 16; bi /= tpr;
    int orow = bi % So; bi /= So;
    int b = bi;
    int t = threadIdx.x;
    int col = ocol0 + (t & 15);
    __shared__ float lds[16 * 132];
    for (int i = 0; i < 8; i++) {
        int c = (t >> 4) + i * 16;
        const float* xp = x + (long)(b * 128 + c) * (S * S);
        const float* wp = dww + c * 9;
        int ir = orow * stride, ic = col * stride;
        float acc = 0.f;
#pragma unroll
        for (int dr = -1; dr <= 1; dr++) {
            int r = ir + dr;
            if (r < 0 || r >= S) continue;
#pragma unroll
            for (int dc = -1; dc <= 1; dc++) {
                int cc = ic + dc;
                if (cc < 0 || cc >= S) continue;
                acc += xp[r * S + cc] * wp[(dr + 1) * 3 + (dc + 1)];
            }
        }
        lds[(t & 15) * 132 + c] = acc;
    }
    __syncthreads();
    if (t < 128) {
        float s = 0.f, s2 = 0.f;
        for (int j = 0; j < 16; j++) { float v = lds[j * 132 + t]; s += v; s2 += v * v; }
        atomicAdd(&stats[t * 2], s);
        atomicAdd(&stats[t * 2 + 1], s2);
    }
    int hwl = t >> 4;
    int cb = (t & 15) * 8;
    long hw = (long)orow * So + ocol0 + hwl;
    v8bf ov;
#pragma unroll
    for (int j = 0; j < 8; j++) ov[j] = (__bf16)lds[hwl * 132 + cb + j];
    *(v8bf*)(out + ((long)b * So * So + hw) * 128 + cb) = ov;
}

// --------------------------- fold BN2d into pointwise weights -> A-fragments
__global__ void prepfold(const float* __restrict__ stats2, const float* __restrict__ bn2g,
                         const float* __restrict__ bn2b, const float* __restrict__ pww,
                         const float* __restrict__ pwb, __bf16* __restrict__ wfrag,
                         float* __restrict__ biasp) {
    int i = blockIdx.x;    // branch
    int t = threadIdx.x;
    __shared__ float sc[128], tc[128];
    float Nel = (i == 0) ? 262144.f : 65536.f;
    if (t < 128) {
        float s = stats2[(i * 128 + t) * 2], s2 = stats2[(i * 128 + t) * 2 + 1];
        float m = s / Nel;
        float var = s2 / Nel - m * m;
        float scl = bn2g[i * 128 + t] * rsqrtf(var + 1e-5f);
        sc[t] = scl;
        tc[t] = bn2b[i * 128 + t] - m * scl;
    }
    __syncthreads();
    if (t < 128) {
        float a = pwb[i * 128 + t];
        const float* wr = pww + (long)(i * 128 + t) * 128;
        for (int c = 0; c < 128; c++) a += wr[c] * tc[c];
        biasp[i * 128 + t] = a;
    }
    for (int j = 0; j < 64; j++) {       // frag layout [mt8][kt4][lane32][e16]
        int idx = t + j * 256;
        int e = idx & 15, lane = (idx >> 4) & 31, kt = (idx >> 9) & 3, mt = (idx >> 11) & 7;
        int half = (lane < 16) ? 0 : 8;
        int r = mt * 16 + (lane & 15);
        int k = kt * 32 + 2 * half + e;
        wfrag[(long)i * 16384 + idx] = (__bf16)(pww[(long)(i * 128 + r) * 128 + k] * sc[k]);
    }
}

// --------------------- pointwise 1x1 (BN-folded) GEMM + GELU, WMMA bf16
// act_in: [row][128] bf16. wfrag: A-fragments of W' (128x128). out: [row][128] bf16.
__global__ __launch_bounds__(256) void pwgemm(const __bf16* __restrict__ act_in,
                                              const __bf16* __restrict__ wfrag,
                                              const float* __restrict__ biasp,
                                              __bf16* __restrict__ act_out) {
    int t = threadIdx.x, wave = t >> 5, lane = t & 31;
    int row = blockIdx.x * 16 + (lane & 15);
    const __bf16* bp = act_in + (long)row * 128 + ((lane >> 4) << 4);
    v8f acc = {0.f, 0.f, 0.f, 0.f, 0.f, 0.f, 0.f, 0.f};
#pragma unroll
    for (int kt = 0; kt < 4; kt++) {
        v16bf bfrag = *(const v16bf*)(bp + kt * 32);
        v16bf afrag = *(const v16bf*)(wfrag + ((wave * 4 + kt) * 32 + lane) * 16);
        acc = __builtin_amdgcn_wmma_f32_16x16x32_bf16(false, afrag, false, bfrag,
                                                      (short)0, acc, false, false);
    }
    int half = (lane < 16) ? 0 : 8;
    int o0 = wave * 16 + half;
    v8bf ov;
#pragma unroll
    for (int r = 0; r < 8; r++) ov[r] = (__bf16)geluf(acc[r] + biasp[o0 + r]);
    *(v8bf*)(act_out + (long)row * 128 + o0) = ov;
}

// ----------------------------- attention: k column softmax stats over n=4096
__global__ void kstats(const __bf16* __restrict__ k_act, float* __restrict__ kmax,
                       float* __restrict__ ksuminv) {
    const int L4 = 4096;
    int bh = blockIdx.x; int b = bh >> 3, h = bh & 7;
    int t = threadIdx.x;
    int d = t & 15, nl = t >> 4;
    const __bf16* kp = k_act + (long)b * L4 * 128 + h * 16 + d;
    float mx = -3.4e38f;
    for (int n = nl; n < L4; n += 16) mx = fmaxf(mx, (float)kp[(long)n * 128]);
    __shared__ float red[256];
    __shared__ float smax[16];
    red[nl * 16 + d] = mx;
    __syncthreads();
    if (t < 16) {
        float m = -3.4e38f;
        for (int j = 0; j < 16; j++) m = fmaxf(m, red[j * 16 + t]);
        smax[t] = m;
    }
    __syncthreads();
    float mxd = smax[d];
    float sum = 0.f;
    for (int n = nl; n < L4; n += 16) sum += expf((float)kp[(long)n * 128] - mxd);
    red[nl * 16 + d] = sum;
    __syncthreads();
    if (t < 16) {
        float s = 0.f;
        for (int j = 0; j < 16; j++) s += red[j * 16 + t];
        kmax[bh * 16 + t] = smax[t];
        ksuminv[bh * 16 + t] = 1.f / s;
    }
}

// kv[b,h,d,e] = sum_n softmax_n(k)[n,d] * v[n,e]
__global__ void kvker(const __bf16* __restrict__ k_act, const __bf16* __restrict__ v_act,
                      const float* __restrict__ kmax, const float* __restrict__ ksuminv,
                      float* __restrict__ kv) {
    const int L4 = 4096;
    int bh = blockIdx.x; int b = bh >> 3, h = bh & 7;
    int t = threadIdx.x;
    int d = t >> 4, e = t & 15;
    __shared__ float sk[256], sv[256], smax[16];
    if (t < 16) smax[t] = kmax[bh * 16 + t];
    __syncthreads();
    float acc = 0.f;
    int nl = t >> 4, ce = t & 15;
    const __bf16* kb = k_act + (long)b * L4 * 128 + h * 16;
    const __bf16* vb = v_act + (long)b * L4 * 128 + h * 16;
    for (int n0 = 0; n0 < L4; n0 += 16) {
        sk[nl * 16 + ce] = expf((float)kb[(long)(n0 + nl) * 128 + ce] - smax[ce]);
        sv[nl * 16 + ce] = (float)vb[(long)(n0 + nl) * 128 + ce];
        __syncthreads();
#pragma unroll
        for (int j = 0; j < 16; j++) acc += sk[j * 16 + d] * sv[j * 16 + e];
        __syncthreads();
    }
    kv[(long)bh * 256 + d * 16 + e] = acc * ksuminv[bh * 16 + d];
}

// softmax(q over dh) @ kv -> mh [row][c] bf16
__global__ __launch_bounds__(256) void attker(const __bf16* __restrict__ q_act,
                                              const float* __restrict__ kv,
                                              __bf16* __restrict__ mh) {
    const int L = 16384;
    int bi = blockIdx.x;
    int lt = bi & 511; int b = bi >> 9;
    int t = threadIdx.x;
    __shared__ float skv[2048];
    for (int j = t; j < 2048; j += 256) skv[j] = kv[(long)b * 2048 + j];
    __syncthreads();
    int h = t & 7, li = t >> 3;
    long row = (long)b * L + (long)lt * 32 + li;
    v16bf q = *(const v16bf*)(q_act + row * 128 + h * 16);
    float p[16];
    float mx = -3.4e38f;
#pragma unroll
    for (int i = 0; i < 16; i++) { p[i] = (float)q[i]; mx = fmaxf(mx, p[i]); }
    float s = 0.f;
#pragma unroll
    for (int i = 0; i < 16; i++) { p[i] = expf(p[i] - mx); s += p[i]; }
    float inv = 1.f / s;
    const float* kvh = skv + h * 256;
    v16bf o;
#pragma unroll
    for (int e = 0; e < 16; e++) {
        float a = 0.f;
#pragma unroll
        for (int d0 = 0; d0 < 16; d0++) a += p[d0] * kvh[d0 * 16 + e];
        o[e] = (__bf16)(a * inv);
    }
    *(v16bf*)(mh + row * 128 + h * 16) = o;
}

// y[row][c] = wn0*mh + wn1*x (transpose x via LDS) + BN1d stats atomics
__global__ __launch_bounds__(256) void addbn1ker(const float* __restrict__ x,
                                                 const __bf16* __restrict__ mh,
                                                 const float* __restrict__ addw,
                                                 float* __restrict__ y,
                                                 float* __restrict__ stats) {
    int bi = blockIdx.x;
    int lt = bi & 511; bi >>= 9;
    int ct = bi & 3; int b = bi >> 2;
    int c0 = ct * 32, l0 = lt * 32;
    __shared__ float xt[32 * 33];
    __shared__ float rs[8 * 33], rs2[8 * 33];
    int t = threadIdx.x;
    for (int j = 0; j < 4; j++) {
        int idx = t + j * 256;
        int li = idx & 31, ci = idx >> 5;
        xt[ci * 33 + li] = x[(long)(b * 128 + c0 + ci) * 16384 + l0 + li];
    }
    __syncthreads();
    float w0 = fmaxf(addw[0], 0.f), w1 = fmaxf(addw[1], 0.f);
    float inv = 1.f / (w0 + w1 + 1e-12f); w0 *= inv; w1 *= inv;
    int ci = t & 31, lr = t >> 5;
    float ps = 0.f, ps2 = 0.f;
    for (int j = 0; j < 4; j++) {
        int li = lr + j * 8;
        long row = (long)b * 16384 + l0 + li;
        float v = w0 * (float)mh[row * 128 + c0 + ci] + w1 * xt[ci * 33 + li];
        y[row * 128 + c0 + ci] = v;
        ps += v; ps2 += v * v;
    }
    rs[lr * 33 + ci] = ps; rs2[lr * 33 + ci] = ps2;
    __syncthreads();
    if (t < 32) {
        float a = 0.f, a2 = 0.f;
        for (int j = 0; j < 8; j++) { a += rs[j * 33 + t]; a2 += rs2[j * 33 + t]; }
        atomicAdd(&stats[(c0 + t) * 2], a);
        atomicAdd(&stats[(c0 + t) * 2 + 1], a2);
    }
}

__global__ void prepbn1(const float* __restrict__ stats, const float* __restrict__ g,
                        const float* __restrict__ b, float* __restrict__ st) {
    int t = threadIdx.x;
    if (t < 128) {
        const float Nel = 262144.f;
        float m = stats[t * 2] / Nel;
        float var = stats[t * 2 + 1] / Nel - m * m;
        float s = g[t] * rsqrtf(var + 1e-5f);
        st[t * 2] = s;
        st[t * 2 + 1] = b[t] - m * s;
    }
}

__global__ void bnapply(const float* __restrict__ y, const float* __restrict__ st,
                        __bf16* __restrict__ out) {
    long base = ((long)blockIdx.x * 256 + threadIdx.x) * 8;
    int c = (int)(base & 127);
    const float* yp = y + base;
    v8bf ov;
#pragma unroll
    for (int r = 0; r < 8; r++) ov[r] = (__bf16)(yp[r] * st[(c + r) * 2] + st[(c + r) * 2 + 1]);
    *(v8bf*)(out + base) = ov;
}

// ---------------------------------------------------- FFN1: [row,128]@W1 +GELU
__global__ __launch_bounds__(256) void ffn1ker(const __bf16* __restrict__ attout,
                                               const __bf16* __restrict__ w1frag,
                                               const float* __restrict__ b1,
                                               __bf16* __restrict__ fbuf) {
    int t = threadIdx.x, wave = t >> 5, lane = t & 31;
    int mt = wave & 1, nt = wave >> 1;
    int row = blockIdx.x * 64 + nt * 16 + (lane & 15);
    const __bf16* bp = attout + (long)row * 128 + ((lane >> 4) << 4);
    v8f acc = {0.f, 0.f, 0.f, 0.f, 0.f, 0.f, 0.f, 0.f};
#pragma unroll
    for (int kt = 0; kt < 4; kt++) {
        v16bf bfrag = *(const v16bf*)(bp + kt * 32);
        v16bf afrag = *(const v16bf*)(w1frag + ((mt * 4 + kt) * 32 + lane) * 16);
        acc = __builtin_amdgcn_wmma_f32_16x16x32_bf16(false, afrag, false, bfrag,
                                                      (short)0, acc, false, false);
    }
    int half = (lane < 16) ? 0 : 8;
    int o0 = mt * 16 + half;
    v8bf ov;
#pragma unroll
    for (int r = 0; r < 8; r++) ov[r] = (__bf16)geluf(acc[r] + b1[o0 + r]);
    *(v8bf*)(fbuf + (long)row * 32 + o0) = ov;
}

// ------------- FFN2: [row,32]@W2 +GELU, weighted add with attout, BN stats
__global__ __launch_bounds__(256) void ffn2ker(const __bf16* __restrict__ fbuf,
                                               const __bf16* __restrict__ w2frag,
                                               const float* __restrict__ b2,
                                               const __bf16* __restrict__ attout,
                                               const float* __restrict__ addw,
                                               float* __restrict__ z,
                                               float* __restrict__ stats) {
    __shared__ float sstat[256];
    int t = threadIdx.x;
    sstat[t] = 0.f;
    __syncthreads();
    int wave = t >> 5, lane = t & 31;
    int row = blockIdx.x * 16 + (lane & 15);
    v16bf bfrag = *(const v16bf*)(fbuf + (long)row * 32 + ((lane >> 4) << 4));
    v16bf afrag = *(const v16bf*)(w2frag + (wave * 32 + lane) * 16);
    v8f acc = {0.f, 0.f, 0.f, 0.f, 0.f, 0.f, 0.f, 0.f};
    acc = __builtin_amdgcn_wmma_f32_16x16x32_bf16(false, afrag, false, bfrag,
                                                  (short)0, acc, false, false);
    int half = (lane < 16) ? 0 : 8;
    int o0 = wave * 16 + half;
    float w0 = fmaxf(addw[0], 0.f), w1 = fmaxf(addw[1], 0.f);
    float inv = 1.f / (w0 + w1 + 1e-12f); w0 *= inv; w1 *= inv;
    const __bf16* ao = attout + (long)row * 128 + o0;
    float* zp = z + (long)row * 128 + o0;
#pragma unroll
    for (int r = 0; r < 8; r++) {
        float g = geluf(acc[r] + b2[o0 + r]);
        float zv = w0 * g + w1 * (float)ao[r];
        zp[r] = zv;
        atomicAdd(&sstat[(o0 + r) * 2], zv);
        atomicAdd(&sstat[(o0 + r) * 2 + 1], zv * zv);
    }
    __syncthreads();
    atomicAdd(&stats[t], sstat[t]);
}

// -------------------------------- final BN + transpose [row][c] -> [B][C][L]
__global__ __launch_bounds__(256) void finalker(const float* __restrict__ z,
                                                const float* __restrict__ st,
                                                float* __restrict__ out) {
    int bi = blockIdx.x;
    int lt = bi & 511; bi >>= 9;
    int ct = bi & 3; int b = bi >> 2;
    int c0 = ct * 32, l0 = lt * 32;
    __shared__ float zt[32 * 33];
    int t = threadIdx.x;
    for (int j = 0; j < 4; j++) {
        int idx = t + j * 256;
        int ci = idx & 31, li = idx >> 5;
        zt[li * 33 + ci] = z[(long)(b * 16384 + l0 + li) * 128 + c0 + ci];
    }
    __syncthreads();
    int li = t & 31, crow = t >> 5;
    for (int j = 0; j < 4; j++) {
        int ci = crow + j * 8;
        out[(long)(b * 128 + c0 + ci) * 16384 + l0 + li] =
            zt[li * 33 + ci] * st[(c0 + ci) * 2] + st[(c0 + ci) * 2 + 1];
    }
}

// ============================================================ host launcher
extern "C" void kernel_launch(void* const* d_in, const int* in_sizes, int n_in,
                              void* d_out, int out_size, void* d_ws, size_t ws_size,
                              hipStream_t stream) {
    const float* x    = (const float*)d_in[0];
    const float* dww  = (const float*)d_in[1];
    const float* bn2g = (const float*)d_in[2];
    const float* bn2b = (const float*)d_in[3];
    const float* pww  = (const float*)d_in[4];
    const float* pwb  = (const float*)d_in[5];
    const float* addw = (const float*)d_in[6];
    const float* bn1g = (const float*)d_in[7];
    const float* bn1b = (const float*)d_in[8];
    const float* W1   = (const float*)d_in[9];
    const float* b1   = (const float*)d_in[10];
    const float* W2   = (const float*)d_in[11];
    const float* b2   = (const float*)d_in[12];
    const float* faddw = (const float*)d_in[13];
    const float* fbng  = (const float*)d_in[14];
    const float* fbnb  = (const float*)d_in[15];
    float* out = (float*)d_out;

    char* ws = (char*)d_ws;
    size_t off = 0;
    auto alloc = [&](size_t bytes) -> void* {
        void* p = ws + off;
        off += (bytes + 255) & ~(size_t)255;
        return p;
    };
    __bf16* bufA = (__bf16*)alloc(67108864);   // convq -> mh
    __bf16* bufB = (__bf16*)alloc(16777216);   // convk -> fbuf
    __bf16* bufC = (__bf16*)alloc(16777216);   // convv
    __bf16* bufD = (__bf16*)alloc(67108864);   // q_act -> attout
    __bf16* bufE = (__bf16*)alloc(16777216);   // k_act
    __bf16* bufF = (__bf16*)alloc(16777216);   // v_act
    float*  yz   = (float*)alloc(134217728);   // y -> z (in place)
    float*  stats = (float*)alloc(1280 * 4);   // 3*256 bn2 | 256 bn1a | 256 bn1f
    float*  stats2  = stats;
    float*  stats1a = stats + 768;
    float*  stats1f = stats + 1024;
    float*  biasp   = (float*)alloc(384 * 4);
    __bf16* wfragPW = (__bf16*)alloc(3 * 16384 * 2);
    __bf16* w1frag  = (__bf16*)alloc(4096 * 2);
    __bf16* w2frag  = (__bf16*)alloc(4096 * 2);
    float*  kmax    = (float*)alloc(2048 * 4);
    float*  ksuminv = (float*)alloc(2048 * 4);
    float*  kvbuf   = (float*)alloc(32768 * 4);
    float*  st1     = (float*)alloc(256 * 4);
    float*  st2     = (float*)alloc(256 * 4);

    zeroker<<<5, 256, 0, stream>>>(stats, 1280);
    prep_weights<<<1, 256, 0, stream>>>(W1, W2, w1frag, w2frag);

    dwconv<<<16384, 256, 0, stream>>>(x, dww,             bufA, stats2,       1, 128);
    dwconv<<<4096, 256, 0, stream>>>(x, dww + 128 * 9,     bufB, stats2 + 256, 2, 64);
    dwconv<<<4096, 256, 0, stream>>>(x, dww + 2 * 128 * 9, bufC, stats2 + 512, 2, 64);
    prepfold<<<3, 256, 0, stream>>>(stats2, bn2g, bn2b, pww, pwb, wfragPW, biasp);

    pwgemm<<<16384, 256, 0, stream>>>(bufA, wfragPW,          biasp,       bufD);
    pwgemm<<<4096, 256, 0, stream>>>(bufB, wfragPW + 16384,   biasp + 128, bufE);
    pwgemm<<<4096, 256, 0, stream>>>(bufC, wfragPW + 32768,   biasp + 256, bufF);

    kstats<<<128, 256, 0, stream>>>(bufE, kmax, ksuminv);
    kvker<<<128, 256, 0, stream>>>(bufE, bufF, kmax, ksuminv, kvbuf);
    attker<<<8192, 256, 0, stream>>>(bufD, kvbuf, bufA /*mh*/);

    addbn1ker<<<32768, 256, 0, stream>>>(x, bufA, addw, yz, stats1a);
    prepbn1<<<1, 128, 0, stream>>>(stats1a, bn1g, bn1b, st1);
    bnapply<<<16384, 256, 0, stream>>>(yz, st1, bufD /*attout*/);

    ffn1ker<<<4096, 256, 0, stream>>>(bufD, w1frag, b1, bufB /*fbuf*/);
    ffn2ker<<<16384, 256, 0, stream>>>(bufB, w2frag, b2, bufD, faddw, yz /*z*/, stats1f);
    prepbn1<<<1, 128, 0, stream>>>(stats1f, fbng, fbnb, st2);
    finalker<<<32768, 256, 0, stream>>>(yz, st2, out);
}
